// AdaptiveAttention_66571993088602
// MI455X (gfx1250) — compile-verified, matrix operand order checked by reading, not run on hardware
//
#include <hip/hip_runtime.h>
#include <math.h>

// Problem constants (match reference)
#define BB   128
#define NN   49
#define CC   2048
#define DD   512
#define EE   512
#define VOC  10000
#define TT   20
#define DE   1024        // D + E
#define KXH  1536        // vg(512) + we(512) + h(512)
#define NBIG 2560        // 4*D gates + D gate-gt

typedef __attribute__((ext_vector_type(16))) _Float16 v16h;
typedef __attribute__((ext_vector_type(8)))  float    v8f;

// ---------------------------------------------------------------------------
// Fragment loader: builds the CDNA5 16-bit 16x32 WMMA operand layout from a
// row-major f32 matrix X[Rmax, ld].  Per ISA 7.12.2 the per-lane k-offsets are
// {0..7} and {16..23} (+ half*8), i.e. two contiguous 32B runs -> 4x b128
// loads.  Out-of-range rows are CLAMPED (not predicated): a clamped A-row only
// affects D-rows >= M, a clamped B-row only affects D-cols >= N, and both are
// masked at the store, so no branches and no wrong results.
// Requires: ld % 4 == 0, k0 % 32 == 0 (true for every call site).
// ---------------------------------------------------------------------------
__device__ __forceinline__ v16h load_frag_f32(const float* __restrict__ X,
                                              int ld, int row0, int k0, int Rmax) {
  const int lane = threadIdx.x & 31;
  int r = row0 + (lane & 15);
  r = (r < Rmax) ? r : (Rmax - 1);          // clamp, never predicate
  const int half = lane >> 4;
  const float4* base = (const float4*)(X + (size_t)r * ld + k0 + half * 8);
  const float4 q0 = base[0];   // k + 0..3
  const float4 q1 = base[1];   // k + 4..7
  const float4 q2 = base[4];   // k + 16..19
  const float4 q3 = base[5];   // k + 20..23
  union { v16h v; _Float16 h[16]; } f;
  f.h[0]  = (_Float16)q0.x;  f.h[1]  = (_Float16)q0.y;
  f.h[2]  = (_Float16)q0.z;  f.h[3]  = (_Float16)q0.w;
  f.h[4]  = (_Float16)q1.x;  f.h[5]  = (_Float16)q1.y;
  f.h[6]  = (_Float16)q1.z;  f.h[7]  = (_Float16)q1.w;
  f.h[8]  = (_Float16)q2.x;  f.h[9]  = (_Float16)q2.y;
  f.h[10] = (_Float16)q2.z;  f.h[11] = (_Float16)q2.w;
  f.h[12] = (_Float16)q3.x;  f.h[13] = (_Float16)q3.y;
  f.h[14] = (_Float16)q3.z;  f.h[15] = (_Float16)q3.w;
  return f.v;
}

// ---------------------------------------------------------------------------
// Generic WMMA GEMM:  C[M,N] = act( A[M,K] @ W[N,K]^T + bias )
// 4 waves / block; each wave computes a 32x32 tile (2x2 WMMA accumulators).
// k-loop unrolled 2x (all K are multiples of 64) so the scheduler holds two
// independent fragment sets: loads for chunk k+32 overlap WMMAs for chunk k.
// Requires M % 64 == 0 (all call sites: 6272, 128, 2560).
// act: 0 = none, 1 = relu, 2 = tanh
// ---------------------------------------------------------------------------
__global__ void gemm_wmma_kernel(const float* __restrict__ A, int lda,
                                 const float* __restrict__ Bw, int ldb,
                                 const float* __restrict__ bias,
                                 float* __restrict__ C, int ldc,
                                 int M, int Nn, int K, int act) {
  const int wave = threadIdx.x >> 5;
  const int lane = threadIdx.x & 31;
  const int row0 = blockIdx.y * 64 + (wave >> 1) * 32;
  const int col0 = blockIdx.x * 64 + (wave & 1) * 32;

  // Per-lane prefetch base for the weight rows this wave will stream.
  int pfr = col0 + (lane & 15);
  pfr = (pfr < Nn) ? pfr : (Nn - 1);
  const float* pfbase = Bw + (size_t)pfr * ldb;

  v8f zero = {};
  v8f acc00 = zero, acc01 = zero, acc10 = zero, acc11 = zero;

#pragma unroll 2
  for (int k0 = 0; k0 < K; k0 += 32) {
    __builtin_prefetch(pfbase + k0 + 64, 0, 3);   // global_prefetch_b8, 2 chunks ahead

    v16h a0 = load_frag_f32(A,  lda, row0,      k0, M);
    v16h a1 = load_frag_f32(A,  lda, row0 + 16, k0, M);
    v16h b0 = load_frag_f32(Bw, ldb, col0,      k0, Nn);
    v16h b1 = load_frag_f32(Bw, ldb, col0 + 16, k0, Nn);

    acc00 = __builtin_amdgcn_wmma_f32_16x16x32_f16(false, a0, false, b0, (short)0, acc00, false, false);
    acc01 = __builtin_amdgcn_wmma_f32_16x16x32_f16(false, a0, false, b1, (short)0, acc01, false, false);
    acc10 = __builtin_amdgcn_wmma_f32_16x16x32_f16(false, a1, false, b0, (short)0, acc10, false, false);
    acc11 = __builtin_amdgcn_wmma_f32_16x16x32_f16(false, a1, false, b1, (short)0, acc11, false, false);
  }

  const int half = lane >> 4;
  const int cn   = lane & 15;
  v8f accs[2][2];
  accs[0][0] = acc00; accs[0][1] = acc01; accs[1][0] = acc10; accs[1][1] = acc11;
#pragma unroll
  for (int i = 0; i < 2; ++i) {
#pragma unroll
    for (int j = 0; j < 2; ++j) {
      const int col = col0 + j * 16 + cn;
      if (col >= Nn) continue;                     // only N can be ragged
      const float bv = bias ? bias[col] : 0.f;
#pragma unroll
      for (int p = 0; p < 8; ++p) {
        const int row = row0 + i * 16 + half * 8 + p;   // M % 64 == 0 -> in range
        float v = accs[i][j][p] + bv;
        if (act == 1)      v = fmaxf(v, 0.f);
        else if (act == 2) v = tanhf(v);
        C[(size_t)row * ldc + col] = v;
      }
    }
  }
}

// ---------------------------------------------------------------------------
// mean over N axis: mean[b,c] = (1/49) * sum_n img[b,n,c]
// ---------------------------------------------------------------------------
__global__ void mean_kernel(const float* __restrict__ img, float* __restrict__ mean) {
  const int idx = blockIdx.x * blockDim.x + threadIdx.x;  // B*C
  if (idx >= BB * CC) return;
  const int b = idx >> 11, c = idx & (CC - 1);
  const float* p = img + (size_t)b * NN * CC + c;
  float s = 0.f;
  for (int n = 0; n < NN; ++n) s += p[(size_t)n * CC];
  mean[idx] = s * (1.f / 49.f);
}

// ---------------------------------------------------------------------------
// Concatenate recurrent weights once:
//   Wbig[n, 0:1024]    = n<2048 ? Wih[n] : Wx[n-2048]
//   Wbig[n, 1024:1536] = n<2048 ? Whh[n] : Wh2[n-2048]
//   bias_big[n]        = n<2048 ? bih[n]+bhh[n] : 0
// ---------------------------------------------------------------------------
__global__ void build_wbig_kernel(const float* __restrict__ Wih, const float* __restrict__ Whh,
                                  const float* __restrict__ Wx,  const float* __restrict__ Wh2,
                                  const float* __restrict__ bih, const float* __restrict__ bhh,
                                  float* __restrict__ Wbig, float* __restrict__ bias_big) {
  const int idx = blockIdx.x * blockDim.x + threadIdx.x;  // NBIG*KXH
  if (idx >= NBIG * KXH) return;
  const int n = idx / KXH, k = idx % KXH;
  float v;
  if (k < DE) v = (n < 4 * DD) ? Wih[(size_t)n * DE + k] : Wx[(size_t)(n - 4 * DD) * DE + k];
  else {
    const int kk = k - DE;
    v = (n < 4 * DD) ? Whh[(size_t)n * DD + kk] : Wh2[(size_t)(n - 4 * DD) * DD + kk];
  }
  Wbig[idx] = v;
  if (k == 0) bias_big[n] = (n < 4 * DD) ? (bih[n] + bhh[n]) : 0.f;
}

// ---------------------------------------------------------------------------
// xh[b] = [ vg[b] | emb[target[b,t-1]] (0 at t=0) | h[b] ]  -> [B, 1536]
// ---------------------------------------------------------------------------
__global__ void build_xh_kernel(const float* __restrict__ vg, const float* __restrict__ emb,
                                const int* __restrict__ target, const float* __restrict__ h,
                                float* __restrict__ xh, int t) {
  const int idx = blockIdx.x * blockDim.x + threadIdx.x;  // B*KXH
  if (idx >= BB * KXH) return;
  const int b = idx / KXH, j = idx % KXH;
  float v;
  if (j < DD) v = vg[b * DD + j];
  else if (j < DD + EE) {
    v = (t == 0) ? 0.f : emb[(size_t)target[b * TT + (t - 1)] * EE + (j - DD)];
  } else v = h[b * DD + (j - DD - EE)];
  xh[idx] = v;
}

// ---------------------------------------------------------------------------
// LSTM pointwise: pre[b] = [i|f|g|o|gt_pre] (2560 cols)
// ---------------------------------------------------------------------------
__device__ __forceinline__ float sigf(float x) { return 1.f / (1.f + expf(-x)); }

__global__ void lstm_pointwise_kernel(const float* __restrict__ pre, const float* __restrict__ m_in,
                                      float* __restrict__ m_out, float* __restrict__ h_out,
                                      float* __restrict__ cin) {
  const int idx = blockIdx.x * blockDim.x + threadIdx.x;  // B*D
  if (idx >= BB * DD) return;
  const int b = idx >> 9, d = idx & (DD - 1);
  const float* p = pre + (size_t)b * NBIG;
  const float gi = p[d], gf = p[DD + d], gg = p[2 * DD + d], go = p[3 * DD + d], gt = p[4 * DD + d];
  const float m2 = sigf(gf) * m_in[idx] + sigf(gi) * tanhf(gg);
  const float tm = tanhf(m2);
  m_out[idx] = m2;
  h_out[idx] = sigf(go) * tm;
  cin[idx]   = sigf(gt) * tm;
}

// ---------------------------------------------------------------------------
// Attention per batch element (block = 256 threads = 8 wave32)
// ---------------------------------------------------------------------------
__global__ void attention_kernel(const float* __restrict__ H, const float* __restrict__ s,
                                 const float* __restrict__ zbase, const float* __restrict__ Wg,
                                 const float* __restrict__ Ws, const float* __restrict__ wh,
                                 const float* __restrict__ Vf, float* __restrict__ A3,
                                 float* __restrict__ attn_out, int t) {
  __shared__ float shH[DD], shS[DD], shgH[NN], shsWs[NN], shz[NN + 1];
  const int b = blockIdx.x, tid = threadIdx.x;
  const int lane = tid & 31, wave = tid >> 5;

  for (int j = tid; j < DD; j += 256) { shH[j] = H[b * DD + j]; shS[j] = s[b * DD + j]; }
  __syncthreads();

  for (int idx = wave; idx < 2 * NN; idx += 8) {
    const int n = (idx < NN) ? idx : idx - NN;
    const float* w = ((idx < NN) ? Wg : Ws) + (size_t)n * DD;
    const float* v = (idx < NN) ? shH : shS;
    float a = 0.f;
    for (int k = lane; k < DD; k += 32) a += v[k] * w[k];
    for (int o = 16; o; o >>= 1) a += __shfl_xor(a, o, 32);
    if (lane == 0) { if (idx < NN) shgH[n] = a; else shsWs[n] = a; }
  }
  __syncthreads();

  for (int idx = wave; idx < NN + 1; idx += 8) {
    float a = 0.f;
    for (int k = lane; k < NN; k += 32) {
      const float zb = (idx < NN) ? zbase[(size_t)(b * NN + idx) * 64 + k] : shsWs[k];
      a += tanhf(zb + shgH[k]) * wh[k];
    }
    for (int o = 16; o; o >>= 1) a += __shfl_xor(a, o, 32);
    if (lane == 0) shz[idx] = a;
  }
  __syncthreads();

  if (tid == 0) {
    float mx = -3.4e38f;
    for (int i = 0; i < NN + 1; ++i) mx = fmaxf(mx, shz[i]);
    float ss = 0.f;
    for (int i = 0; i < NN + 1; ++i) { float e = expf(shz[i] - mx); shz[i] = e; ss += e; }
    const float inv = 1.f / ss;
    for (int i = 0; i < NN + 1; ++i) shz[i] *= inv;
  }
  __syncthreads();

  if (tid < NN) attn_out[((size_t)b * TT + t) * NN + tid] = shz[tid];

  for (int d = tid; d < DD; d += 256) {
    float c = shz[NN] * shS[d];
    for (int n = 0; n < NN; ++n) c += shz[n] * Vf[(size_t)(b * NN + n) * DD + d];
    A3[b * DD + d] = shH[d] + c;
  }
}

// ---------------------------------------------------------------------------
// In-place log_softmax over the last axis (one block per row of 10000).
// ---------------------------------------------------------------------------
__global__ void logsoftmax_kernel(float* __restrict__ logits, int cols) {
  __shared__ float red[256];
  const int r = blockIdx.x, tid = threadIdx.x;
  float* p = logits + (size_t)r * cols;

  float m = -3.4e38f;
  for (int j = tid; j < cols; j += 256) m = fmaxf(m, p[j]);
  red[tid] = m; __syncthreads();
  for (int s = 128; s; s >>= 1) { if (tid < s) red[tid] = fmaxf(red[tid], red[tid + s]); __syncthreads(); }
  m = red[0]; __syncthreads();

  float ss = 0.f;
  for (int j = tid; j < cols; j += 256) ss += expf(p[j] - m);
  red[tid] = ss; __syncthreads();
  for (int s = 128; s; s >>= 1) { if (tid < s) red[tid] += red[tid + s]; __syncthreads(); }
  const float lse = m + logf(red[0]); __syncthreads();

  for (int j = tid; j < cols; j += 256) p[j] = p[j] - lse;
}

// ---------------------------------------------------------------------------
// Host-side GEMM launcher
// ---------------------------------------------------------------------------
static void gemm(hipStream_t s, const float* A, int lda, const float* Bw, int ldb,
                 const float* bias, float* C, int ldc, int M, int N, int K, int act) {
  dim3 grid((N + 63) / 64, (M + 63) / 64), block(128);
  gemm_wmma_kernel<<<grid, block, 0, s>>>(A, lda, Bw, ldb, bias, C, ldc, M, N, K, act);
}

extern "C" void kernel_launch(void* const* d_in, const int* in_sizes, int n_in,
                              void* d_out, int out_size, void* d_ws, size_t ws_size,
                              hipStream_t stream) {
  (void)in_sizes; (void)n_in; (void)out_size; (void)ws_size;

  const float* img    = (const float*)d_in[0];
  const int*   target = (const int*)  d_in[1];
  const float* emb    = (const float*)d_in[2];
  const float* Wa  = (const float*)d_in[3];  const float* ba  = (const float*)d_in[4];
  const float* Wb  = (const float*)d_in[5];  const float* bb  = (const float*)d_in[6];
  const float* Whi = (const float*)d_in[7];  const float* bhi = (const float*)d_in[8];
  const float* Wmi = (const float*)d_in[9];  const float* bmi = (const float*)d_in[10];
  const float* Wih = (const float*)d_in[11]; const float* bih = (const float*)d_in[12];
  const float* Whh = (const float*)d_in[13]; const float* bhh = (const float*)d_in[14];
  const float* Wv  = (const float*)d_in[15];
  const float* Wg  = (const float*)d_in[16];
  const float* wh  = (const float*)d_in[17];
  const float* WHm = (const float*)d_in[18];
  const float* Wx  = (const float*)d_in[19];
  const float* Wh2 = (const float*)d_in[20];
  const float* Ws  = (const float*)d_in[21];
  const float* Wc  = (const float*)d_in[22]; const float* bc  = (const float*)d_in[23];
  const float* Wfc = (const float*)d_in[24]; const float* bfc = (const float*)d_in[25];
  const float* Wp  = (const float*)d_in[26]; const float* bp  = (const float*)d_in[27];

  // Workspace carve-up (floats)
  float* ws = (float*)d_ws;
  float* Vf      = ws; ws += (size_t)BB * NN * DD;      // 3,211,264
  float* meanb   = ws; ws += (size_t)BB * CC;           //   262,144
  float* zbase   = ws; ws += (size_t)BB * NN * 64;      //   401,408 (ld 64, cols<49 valid)
  float* vg      = ws; ws += (size_t)BB * DD;
  float* hbuf0   = ws; ws += (size_t)BB * DD;
  float* hbuf1   = ws; ws += (size_t)BB * DD;
  float* mbuf0   = ws; ws += (size_t)BB * DD;
  float* mbuf1   = ws; ws += (size_t)BB * DD;
  float* xh      = ws; ws += (size_t)BB * KXH;
  float* pre     = ws; ws += (size_t)BB * NBIG;
  float* cin     = ws; ws += (size_t)BB * DD;
  float* Hb      = ws; ws += (size_t)BB * DD;
  float* sb      = ws; ws += (size_t)BB * DD;
  float* A3      = ws; ws += (size_t)BB * DD;
  float* out_all = ws; ws += (size_t)BB * TT * DD;      // rows r = b*T+t
  float* Wbig    = ws; ws += (size_t)NBIG * KXH;        // 3,932,160
  float* biasbig = ws; ws += (size_t)NBIG;

  float* logits = (float*)d_out;                          // [B*T, VOC], r = b*T+t
  float* attn   = (float*)d_out + (size_t)BB * TT * VOC;  // [B, T, N]

  float* hcur[2] = { hbuf0, hbuf1 };
  float* mcur[2] = { mbuf0, mbuf1 };

  // ---- setup (once per call) ----
  mean_kernel<<<(BB * CC) / 256, 256, 0, stream>>>(img, meanb);
  build_wbig_kernel<<<(NBIG * KXH + 255) / 256, 256, 0, stream>>>(Wih, Whh, Wx, Wh2, bih, bhh, Wbig, biasbig);

  // Vf = relu(img @ Wa^T + ba)             [6272, 512]
  gemm(stream, img,   CC, Wa,  CC, ba,  Vf,    DD, BB * NN, DD, CC, 1);
  // vg/h0/m0 = relu(mean @ W^T + b)        [128, 512]
  gemm(stream, meanb, CC, Wb,  CC, bb,  vg,      DD, BB, DD, CC, 1);
  gemm(stream, meanb, CC, Whi, CC, bhi, hcur[0], DD, BB, DD, CC, 1);
  gemm(stream, meanb, CC, Wmi, CC, bmi, mcur[0], DD, BB, DD, CC, 1);
  // zbase = Vf @ Wv^T                      [6272, 49] (ld 64)
  gemm(stream, Vf, DD, Wv, DD, nullptr, zbase, 64, BB * NN, NN, DD, 0);

  // ---- recurrent scan ----
  int cur = 0;
  for (int t = 0; t < TT; ++t) {
    const int nxt = cur ^ 1;
    build_xh_kernel<<<(BB * KXH) / 256, 256, 0, stream>>>(vg, emb, target, hcur[cur], xh, t);
    // pre = xh @ Wbig^T + biasbig          [128, 2560]  (gates + gt_pre fused)
    gemm(stream, xh, KXH, Wbig, KXH, biasbig, pre, NBIG, BB, NBIG, KXH, 0);
    lstm_pointwise_kernel<<<(BB * DD) / 256, 256, 0, stream>>>(pre, mcur[cur], mcur[nxt], hcur[nxt], cin);
    // H = relu(h2 @ WH^T); s = relu(cin @ Wc^T + bc)
    gemm(stream, hcur[nxt], DD, WHm, DD, nullptr, Hb, DD, BB, DD, DD, 1);
    gemm(stream, cin,       DD, Wc,  DD, bc,      sb, DD, BB, DD, DD, 1);
    attention_kernel<<<BB, 256, 0, stream>>>(Hb, sb, zbase, Wg, Ws, wh, Vf, A3, attn, t);
    // out[:,t] = tanh(A3 @ Wfc^T + bfc)  -> out_all rows b*T+t (strided store)
    gemm(stream, A3, DD, Wfc, DD, bfc, out_all + (size_t)t * DD, TT * DD, BB, DD, DD, 2);
    cur = nxt;
  }

  // ---- deferred vocabulary projection over all (b,t) rows + log_softmax ----
  gemm(stream, out_all, DD, Wp, DD, bp, logits, VOC, BB * TT, VOC, DD, 0);
  logsoftmax_kernel<<<BB * TT, 256, 0, stream>>>(logits, VOC);
}
